// Perceiver_46471546143524
// MI455X (gfx1250) — matrix operand order repaired and checked by reading, not compiled, not run on hardware
//
#include <hip/hip_runtime.h>
#include <math.h>

// ---------------- constants (match reference) ----------------
#define CB    4        // batch
#define CDS   8192     // data tokens
#define CDC   64       // data channels
#define CLS   256      // latents
#define CLC   128      // latent channels
#define CH    8        // heads
#define CD    16       // head dim
#define CINNER 128     // H*D
#define CFFH  16384    // FF_HID = LC*LC
#define NTOK  (CB*CLS) // 1024 latent tokens

typedef __attribute__((ext_vector_type(16))) _Float16 v16h;
typedef __attribute__((ext_vector_type(8)))  float    v8f;

// ---------- WMMA fragment index helpers (contiguous k order used by this file) ----------
// A-matrix 16x32 (MxK): lane holds row M=lane%16; element e -> K:
//   K(e,lane) = e + (lane<16?0:16)              (one contiguous 16-run)
// B-matrix 32x16 (KxN): lane holds col N=lane%16; element e -> K:
//   K(e,lane) = e + (lane<16?0:16)              (one contiguous 16-run)
__device__ __forceinline__ int a_kidx(int e, int lane) {
  return e + ((lane < 16) ? 0 : 16);
}

__device__ __forceinline__ float gelu_exact(float x) {
  return 0.5f * x * (1.0f + erff(x * 0.70710678118654752f));
}

// Pack two f32 quads (one contiguous 8-run) into fragment elements [e0..e0+7]
__device__ __forceinline__ void pack8(v16h& f, int e0, float4 a, float4 b) {
  f[e0 + 0] = (_Float16)a.x; f[e0 + 1] = (_Float16)a.y;
  f[e0 + 2] = (_Float16)a.z; f[e0 + 3] = (_Float16)a.w;
  f[e0 + 4] = (_Float16)b.x; f[e0 + 5] = (_Float16)b.y;
  f[e0 + 6] = (_Float16)b.z; f[e0 + 7] = (_Float16)b.w;
}

// Load f32 A fragment (row-major, K mult of 32) as two vectorized 8-runs
__device__ __forceinline__ v16h load_a_frag(const float* __restrict__ A,
                                            size_t rowBase, int kb, int lane) {
  int h16 = (lane < 16) ? 0 : 16;
  const float4* p0 = (const float4*)(A + rowBase + kb + h16);
  const float4* p1 = (const float4*)(A + rowBase + kb + h16 + 8);
  v16h f;
  pack8(f, 0, p0[0], p0[1]);
  pack8(f, 8, p1[0], p1[1]);
  return f;
}

// =====================================================================
// Weight pre-pass: Wt[n*K+k] = (f16) W[k*N+n]   (B-fragment-native layout)
// =====================================================================
__global__ void w_to_f16t(const float* __restrict__ W, _Float16* __restrict__ Wt,
                          int K, int N) {
  int i = blockIdx.x * blockDim.x + threadIdx.x;
  if (i < K * N) {
    int n = i / K, k = i - n * K;
    Wt[i] = (_Float16)W[(size_t)k * N + n];
  }
}

// =====================================================================
// GEMM: C[M,N] = A[M,K] @ W[K,N] (+bias).  A fp32 row-major, W f16 [N][K].
// omode: 0 = f32 [M,N]; 1 = f16 [M,N]; 2 = f16 transposed [M/Sa][N][Sa].
// One wave per 16x16 output tile; K multiple of 32.
// =====================================================================
__global__ void gemm_wmma(const float* __restrict__ A, const _Float16* __restrict__ Wt,
                          const float* __restrict__ bias, void* __restrict__ Cv,
                          int M, int K, int N, int omode, int Sa) {
  int lane = threadIdx.x & 31;
  int wave = (int)((blockIdx.x * blockDim.x + threadIdx.x) >> 5);
  int ntn = N >> 4;
  int tiles = (M >> 4) * ntn;
  if (wave >= tiles) return;                 // wave-uniform exit (EXEC all-ones)
  int tm = wave / ntn, tn = wave - tm * ntn;
  size_t arowBase = (size_t)(tm * 16 + (lane & 15)) * K;
  int coln = tn * 16 + (lane & 15);
  int h16 = (lane < 16) ? 0 : 16;
  const _Float16* wcol = Wt + (size_t)coln * K + h16;

  v8f acc;
  float bv = bias ? bias[coln] : 0.0f;
#pragma unroll
  for (int i = 0; i < 8; ++i) acc[i] = bv;

  for (int kb = 0; kb < K; kb += 32) {
    __builtin_prefetch(wcol + kb + 64, 0, 1);   // global_prefetch_b8 (next tiles)
    v16h a = load_a_frag(A, arowBase, kb, lane);
    v16h b = *(const v16h*)(wcol + kb);         // 16 contiguous f16, 32B aligned
    acc = __builtin_amdgcn_wmma_f32_16x16x32_f16(false, a, false, b, (short)0, acc,
                                                 false, false);
  }

  int mb = tm * 16 + ((lane < 16) ? 0 : 8);
  if (omode == 0) {
    float* C = (float*)Cv;
#pragma unroll
    for (int v = 0; v < 8; ++v) C[(size_t)(mb + v) * N + coln] = acc[v];
  } else if (omode == 1) {
    _Float16* C = (_Float16*)Cv;
#pragma unroll
    for (int v = 0; v < 8; ++v) C[(size_t)(mb + v) * N + coln] = (_Float16)acc[v];
  } else {
    _Float16* C = (_Float16*)Cv;                // [b][n][s], s = m % Sa
#pragma unroll
    for (int v = 0; v < 8; ++v) {
      int m = mb + v;
      C[((size_t)(m / Sa) * N + coln) * Sa + (m % Sa)] = (_Float16)acc[v];
    }
  }
}

// =====================================================================
// Flash attention: one wave per (b, h, latent-16-tile).  blockDim = 32.
// Q f32 [B,LS,INNER]; Kf f16 [B,S,INNER]; Vt f16 [B][INNER][S].
// logits K-dim D=16 zero-padded to 32.
// =====================================================================
__global__ void attn_wmma(const float* __restrict__ Q, const _Float16* __restrict__ Kf,
                          const _Float16* __restrict__ Vt, float* __restrict__ O,
                          int S) {
  const float scale = 0.25f;  // D^-0.5
  int lane = threadIdx.x;
  int bid  = blockIdx.x;
  int lt = bid & 15;           // LS/16 tiles
  int h  = (bid >> 4) & 7;     // 8 heads
  int b  = bid >> 7;           // batch

  __shared__ _Float16 plds[16][32];

  // Q tile as A fragment (16 latents x K=32; d>=16 zero), pre-scaled.
  v16h qa;
  {
    int m = lt * 16 + (lane & 15);
    const float4* qp =
        (const float4*)(Q + (size_t)(b * CLS + m) * CINNER + h * CD);
    float4 q0 = qp[0], q1 = qp[1], q2 = qp[2], q3 = qp[3];
    q0.x *= scale; q0.y *= scale; q0.z *= scale; q0.w *= scale;
    q1.x *= scale; q1.y *= scale; q1.z *= scale; q1.w *= scale;
    q2.x *= scale; q2.y *= scale; q2.z *= scale; q2.w *= scale;
    q3.x *= scale; q3.y *= scale; q3.z *= scale; q3.w *= scale;
    pack8(qa, 0, q0, q1);
    pack8(qa, 8, q2, q3);
#pragma unroll
    for (int e = 0; e < 16; ++e) if (lane >= 16) qa[e] = (_Float16)0.0f;   // padded d>=16
  }

  v16h zv;
#pragma unroll
  for (int e = 0; e < 16; ++e) zv[e] = (_Float16)0.0f;

  float rmax[8], rsum[8];
  v8f oacc;
#pragma unroll
  for (int v = 0; v < 8; ++v) { rmax[v] = -INFINITY; rsum[v] = 0.0f; oacc[v] = 0.0f; }

  const _Float16* Vrow = Vt + ((size_t)b * CINNER + h * CD + (lane & 15)) * S
                         + ((lane < 16) ? 0 : 16);

  for (int s0 = 0; s0 < S; s0 += 32) {
    // ---- logits for 32 data tokens (B = K^T, d padded; lanes>=16 -> d>=16 -> 0)
    size_t tok0 = (size_t)(b * S + s0 + (lane & 15));
    v16h kr0 = *(const v16h*)(Kf + tok0 * CINNER + h * CD);
    v16h kr1 = *(const v16h*)(Kf + (tok0 + 16) * CINNER + h * CD);
    v16h kb0 = (lane < 16) ? kr0 : zv;
    v16h kb1 = (lane < 16) ? kr1 : zv;

    v8f zc;
#pragma unroll
    for (int i = 0; i < 8; ++i) zc[i] = 0.0f;
    v8f l0 = __builtin_amdgcn_wmma_f32_16x16x32_f16(false, qa, false, kb0, (short)0, zc,
                                                    false, false);
    v8f l1 = __builtin_amdgcn_wmma_f32_16x16x32_f16(false, qa, false, kb1, (short)0, zc,
                                                    false, false);

    // ---- online softmax (rows live in 16-lane halves) ----
#pragma unroll
    for (int v = 0; v < 8; ++v) {
      float bm = fmaxf(l0[v], l1[v]);
#pragma unroll
      for (int off = 1; off < 16; off <<= 1) bm = fmaxf(bm, __shfl_xor(bm, off, 32));
      float mnew = fmaxf(rmax[v], bm);
      float corr = __expf(rmax[v] - mnew);
      float p0 = __expf(l0[v] - mnew);
      float p1 = __expf(l1[v] - mnew);
      float ps = p0 + p1;
#pragma unroll
      for (int off = 1; off < 16; off <<= 1) ps += __shfl_xor(ps, off, 32);
      rsum[v] = rsum[v] * corr + ps;
      rmax[v] = mnew;
      oacc[v] *= corr;
      l0[v] = p0; l1[v] = p1;
    }

    // ---- transpose P (C layout -> A layout) via LDS ----
    __syncthreads();
    int crow = (lane < 16) ? 0 : 8;
#pragma unroll
    for (int v = 0; v < 8; ++v) {
      plds[crow + v][lane & 15]        = (_Float16)l0[v];
      plds[crow + v][16 + (lane & 15)] = (_Float16)l1[v];
    }
    __syncthreads();
    v16h pa;
#pragma unroll
    for (int e = 0; e < 16; ++e) pa[e] = plds[lane & 15][a_kidx(e, lane)];

    // ---- O += P @ Vtile  (K=32 over tokens; Vt gives 16 contiguous f16) ----
    v16h vbf = *(const v16h*)(Vrow + s0);
    oacc = __builtin_amdgcn_wmma_f32_16x16x32_f16(false, pa, false, vbf, (short)0, oacc,
                                                  false, false);
  }

#pragma unroll
  for (int v = 0; v < 8; ++v) {
    int m = lt * 16 + ((lane < 16) ? 0 : 8) + v;
    O[((size_t)(b * CLS + m)) * CINNER + h * CD + (lane & 15)] = oacc[v] / rsum[v];
  }
}

// =====================================================================
// Fused GLU-FFW: out += (xn@W1_lat+b1)*gelu(xn@W1_gate+b1g) @ W2
// W1t f16 [2*FFH][LC]; W2t f16 [LC][FFH].
// grid.x = token tiles, grid.y = chunk groups; blockDim = 32.
// =====================================================================
__global__ void ffw_fused(const float* __restrict__ xn, const _Float16* __restrict__ W1t,
                          const float* __restrict__ b1, const _Float16* __restrict__ W2t,
                          float* __restrict__ out, int chunksPerGroup) {
  int lane = threadIdx.x;
  int tt  = blockIdx.x;
  int grp = blockIdx.y;
  int nc  = lane & 15;
  int h16 = (lane < 16) ? 0 : 16;
  __shared__ _Float16 hlds[16][32];

  // xn tile as 4 A fragments (K=128), vectorized loads
  v16h xa[4];
  {
    size_t rowBase = (size_t)(tt * 16 + nc) * CLC;
#pragma unroll
    for (int kb = 0; kb < 4; ++kb) xa[kb] = load_a_frag(xn, rowBase, kb * 32, lane);
  }

  v8f acc[8];
#pragma unroll
  for (int nt = 0; nt < 8; ++nt)
#pragma unroll
    for (int i = 0; i < 8; ++i) acc[nt][i] = 0.0f;

  int c0 = grp * chunksPerGroup;
  for (int ch = c0; ch < c0 + chunksPerGroup; ++ch) {
    int f0 = ch * 32;
    // W1t rows for this chunk's 32 hidden columns (lat + gate halves)
    const _Float16* w1l0 = W1t + (size_t)(f0 + nc) * CLC + h16;
    const _Float16* w1l1 = W1t + (size_t)(f0 + 16 + nc) * CLC + h16;
    const _Float16* w1g0 = W1t + (size_t)(CFFH + f0 + nc) * CLC + h16;
    const _Float16* w1g1 = W1t + (size_t)(CFFH + f0 + 16 + nc) * CLC + h16;
    __builtin_prefetch(w1l0 + 32 * CLC, 0, 1);      // next chunk (speculative)
    __builtin_prefetch(w1g0 + 32 * CLC, 0, 1);

    v8f ul0, ul1, ug0, ug1;
    {
      float bl0 = b1[f0 + nc], bl1 = b1[f0 + 16 + nc];
      float bg0 = b1[CFFH + f0 + nc], bg1 = b1[CFFH + f0 + 16 + nc];
#pragma unroll
      for (int i = 0; i < 8; ++i) { ul0[i] = bl0; ul1[i] = bl1; ug0[i] = bg0; ug1[i] = bg1; }
    }
#pragma unroll
    for (int kb = 0; kb < 4; ++kb) {
      v16h w0 = *(const v16h*)(w1l0 + kb * 32);
      v16h w1 = *(const v16h*)(w1l1 + kb * 32);
      v16h g0 = *(const v16h*)(w1g0 + kb * 32);
      v16h g1 = *(const v16h*)(w1g1 + kb * 32);
      ul0 = __builtin_amdgcn_wmma_f32_16x16x32_f16(false, xa[kb], false, w0, (short)0, ul0, false, false);
      ul1 = __builtin_amdgcn_wmma_f32_16x16x32_f16(false, xa[kb], false, w1, (short)0, ul1, false, false);
      ug0 = __builtin_amdgcn_wmma_f32_16x16x32_f16(false, xa[kb], false, g0, (short)0, ug0, false, false);
      ug1 = __builtin_amdgcn_wmma_f32_16x16x32_f16(false, xa[kb], false, g1, (short)0, ug1, false, false);
    }

    // h = lat * gelu(gate); stage through LDS to reach A layout
    __syncthreads();
    int crow = (lane < 16) ? 0 : 8;
#pragma unroll
    for (int v = 0; v < 8; ++v) {
      hlds[crow + v][nc]      = (_Float16)(ul0[v] * gelu_exact(ug0[v]));
      hlds[crow + v][16 + nc] = (_Float16)(ul1[v] * gelu_exact(ug1[v]));
    }
    __syncthreads();
    v16h ha;
#pragma unroll
    for (int e = 0; e < 16; ++e) ha[e] = hlds[nc][a_kidx(e, lane)];

#pragma unroll
    for (int nt = 0; nt < 8; ++nt) {
      v16h w2t = *(const v16h*)(W2t + (size_t)(nt * 16 + nc) * CFFH + f0 + h16);
      acc[nt] = __builtin_amdgcn_wmma_f32_16x16x32_f16(false, ha, false, w2t, (short)0,
                                                       acc[nt], false, false);
    }
  }

  int crow = (lane < 16) ? 0 : 8;
#pragma unroll
  for (int nt = 0; nt < 8; ++nt)
#pragma unroll
    for (int v = 0; v < 8; ++v)
      atomicAdd(&out[(size_t)(tt * 16 + crow + v) * CLC + nt * 16 + nc], acc[nt][v]);
}

// =====================================================================
// Small elementwise / reduction kernels
// =====================================================================
__global__ void ln_rows(const float* __restrict__ x, const float* __restrict__ g,
                        const float* __restrict__ bta, float* __restrict__ xn, int nrows) {
  int wave = (int)((blockIdx.x * blockDim.x + threadIdx.x) >> 5);
  int lane = threadIdx.x & 31;
  if (wave >= nrows) return;
  const float4* xr = (const float4*)(x + (size_t)wave * CLC);
  float4 q = xr[lane];
  float s = q.x + q.y + q.z + q.w;
#pragma unroll
  for (int off = 1; off < 32; off <<= 1) s += __shfl_xor(s, off, 32);
  float mu = s * (1.0f / CLC);
  float dx = q.x - mu, dy = q.y - mu, dz = q.z - mu, dw = q.w - mu;
  float sv = dx * dx + dy * dy + dz * dz + dw * dw;
#pragma unroll
  for (int off = 1; off < 32; off <<= 1) sv += __shfl_xor(sv, off, 32);
  float r = rsqrtf(sv * (1.0f / CLC) + 1e-5f);
  int c = lane * 4;
  float* o = xn + (size_t)wave * CLC + c;
  o[0] = dx * r * g[c + 0] + bta[c + 0];
  o[1] = dy * r * g[c + 1] + bta[c + 1];
  o[2] = dz * r * g[c + 2] + bta[c + 2];
  o[3] = dw * r * g[c + 3] + bta[c + 3];
}

__global__ void bcast_latents(const float* __restrict__ src, float* __restrict__ dst) {
  int i = blockIdx.x * blockDim.x + threadIdx.x;
  if (i < NTOK * CLC) dst[i] = src[i % (CLS * CLC)];
}
__global__ void avg2(const float* __restrict__ a, const float* __restrict__ b,
                     float* __restrict__ o, int n) {
  int i = blockIdx.x * blockDim.x + threadIdx.x;
  if (i < n) o[i] = 0.5f * (a[i] + b[i]);
}
__global__ void copyk(const float* __restrict__ a, float* __restrict__ o, int n) {
  int i = blockIdx.x * blockDim.x + threadIdx.x;
  if (i < n) o[i] = a[i];
}
__global__ void resbias(const float* __restrict__ x, const float* __restrict__ b2,
                        float* __restrict__ o, int n) {
  int i = blockIdx.x * blockDim.x + threadIdx.x;
  if (i < n) o[i] = x[i] + b2[i & (CLC - 1)];
}

// =====================================================================
// Host orchestration
// =====================================================================
extern "C" void kernel_launch(void* const* d_in, const int* in_sizes, int n_in,
                              void* d_out, int out_size, void* d_ws, size_t ws_size,
                              hipStream_t stream) {
  const float* data     = (const float*)d_in[0];
  const float* init_lat = (const float*)d_in[1];
  const float* ca_Wq = (const float*)d_in[2];
  const float* ca_Wk = (const float*)d_in[3];
  const float* ca_Wv = (const float*)d_in[4];
  const float* ca_Wo = (const float*)d_in[5];
  const float* ca_bo = (const float*)d_in[6];
  const float* tr_Wq = (const float*)d_in[7];
  const float* tr_Wk = (const float*)d_in[8];
  const float* tr_Wv = (const float*)d_in[9];
  const float* tr_Wo = (const float*)d_in[10];
  const float* tr_bo = (const float*)d_in[11];
  const float* caff_g  = (const float*)d_in[12];
  const float* caff_b  = (const float*)d_in[13];
  const float* caff_W1 = (const float*)d_in[14];
  const float* caff_b1 = (const float*)d_in[15];
  const float* caff_W2 = (const float*)d_in[16];
  const float* caff_b2 = (const float*)d_in[17];
  const float* trff_g  = (const float*)d_in[18];
  const float* trff_b  = (const float*)d_in[19];
  const float* trff_W1 = (const float*)d_in[20];
  const float* trff_b1 = (const float*)d_in[21];
  const float* trff_W2 = (const float*)d_in[22];
  const float* trff_b2 = (const float*)d_in[23];

  const int NLAT = NTOK * CLC;           // 131072

  // ---- f32 scratch ----
  float* f32w = (float*)d_ws;
  float* lat = f32w;                 float* zb  = lat + NLAT;
  float* qb  = zb  + NLAT;           float* oat = qb + NLAT;
  float* prj = oat + NLAT;           float* xnb = prj + NLAT;
  float* ffo = xnb + NLAT;

  // ---- f16 scratch ----
  _Float16* h16 = (_Float16*)(ffo + NLAT);
  _Float16* Kc16   = h16;                                  // [B,DS,INNER]
  _Float16* Vt16   = Kc16 + (size_t)CB * CDS * CINNER;     // [B][INNER][DS]
  _Float16* kk16   = Vt16 + (size_t)CB * CDS * CINNER;     // [B,LS,INNER]
  _Float16* vvt16  = kk16 + (size_t)NTOK * CINNER;         // [B][INNER][LS]
  _Float16* caWq_t = vvt16 + (size_t)NTOK * CINNER;        // [INNER][LC]
  _Float16* caWk_t = caWq_t + CLC * CINNER;                // [INNER][DC]
  _Float16* caWv_t = caWk_t + CDC * CINNER;
  _Float16* caWo_t = caWv_t + CDC * CINNER;                // [LC][INNER]
  _Float16* trWq_t = caWo_t + CINNER * CLC;                // [2][INNER][LC]
  _Float16* trWk_t = trWq_t + 2 * CLC * CINNER;
  _Float16* trWv_t = trWk_t + 2 * CLC * CINNER;
  _Float16* trWo_t = trWv_t + 2 * CLC * CINNER;
  _Float16* caW1_t = trWo_t + 2 * CINNER * CLC;            // [2*FFH][LC]
  _Float16* caW2_t = caW1_t + (size_t)CLC * 2 * CFFH;      // [LC][FFH]
  _Float16* trW1_t = caW2_t + (size_t)CFFH * CLC;
  _Float16* trW2_t = trW1_t + (size_t)CLC * 2 * CFFH;

  auto wconv = [&](const float* W, _Float16* Wt, int K, int N) {
    int n = K * N;
    w_to_f16t<<<(n + 255) / 256, 256, 0, stream>>>(W, Wt, K, N);
  };
  auto gemm = [&](const float* A, const _Float16* Wt, const float* bias, void* C,
                  int M, int K, int N, int omode, int Sa) {
    int tiles = (M / 16) * (N / 16);
    gemm_wmma<<<(tiles + 3) / 4, 128, 0, stream>>>(A, Wt, bias, C, M, K, N, omode, Sa);
  };
  auto ew = [&](int n) { return dim3((n + 255) / 256); };

  // ---- weight conversion pre-pass (runs every call; deterministic) ----
  wconv(ca_Wq, caWq_t, CLC, CINNER);
  wconv(ca_Wk, caWk_t, CDC, CINNER);
  wconv(ca_Wv, caWv_t, CDC, CINNER);
  wconv(ca_Wo, caWo_t, CINNER, CLC);
  for (int t = 0; t < 2; ++t) {
    wconv(tr_Wq + (size_t)t * CLC * CINNER, trWq_t + (size_t)t * CLC * CINNER, CLC, CINNER);
    wconv(tr_Wk + (size_t)t * CLC * CINNER, trWk_t + (size_t)t * CLC * CINNER, CLC, CINNER);
    wconv(tr_Wv + (size_t)t * CLC * CINNER, trWv_t + (size_t)t * CLC * CINNER, CLC, CINNER);
    wconv(tr_Wo + (size_t)t * CINNER * CLC, trWo_t + (size_t)t * CINNER * CLC, CINNER, CLC);
  }
  wconv(caff_W1, caW1_t, CLC, 2 * CFFH);
  wconv(caff_W2, caW2_t, CFFH, CLC);
  wconv(trff_W1, trW1_t, CLC, 2 * CFFH);
  wconv(trff_W2, trW2_t, CFFH, CLC);

  // ---- data K/V projections: computed ONCE (ca params shared, data constant) ----
  gemm(data, caWk_t, nullptr, Kc16, CB * CDS, CDC, CINNER, 1, 0);
  gemm(data, caWv_t, nullptr, Vt16, CB * CDS, CDC, CINNER, 2, CDS);

  // ---- latents = broadcast(init_latents) ----
  bcast_latents<<<ew(NLAT), 256, 0, stream>>>(init_lat, lat);

  auto cross_attn = [&](float* dst) {            // dst = attn(data, lat) + ca_bo
    gemm(lat, caWq_t, nullptr, qb, NTOK, CLC, CINNER, 0, 0);
    attn_wmma<<<CB * CH * (CLS / 16), 32, 0, stream>>>(qb, Kc16, Vt16, oat, CDS);
    gemm(oat, caWo_t, ca_bo, dst, NTOK, CINNER, CLC, 0, 0);
  };
  auto self_attn = [&](float* x, int t) {        // x = attn(x, x; tr params[t])
    size_t wof = (size_t)t * CLC * CINNER;
    gemm(x, trWq_t + wof, nullptr, qb,    NTOK, CLC, CINNER, 0, 0);
    gemm(x, trWk_t + wof, nullptr, kk16,  NTOK, CLC, CINNER, 1, 0);
    gemm(x, trWv_t + wof, nullptr, vvt16, NTOK, CLC, CINNER, 2, CLS);
    attn_wmma<<<CB * CH * (CLS / 16), 32, 0, stream>>>(qb, kk16, vvt16, oat, CLS);
    gemm(oat, trWo_t + wof, tr_bo + (size_t)t * CLC, x, NTOK, CINNER, CLC, 0, 0);
  };
  auto ffw = [&](const float* g, const float* bta, const _Float16* W1t,
                 const float* b1, const _Float16* W2t, const float* b2) {
    ln_rows<<<NTOK / 4, 128, 0, stream>>>(lat, g, bta, xnb, NTOK);
    resbias<<<ew(NLAT), 256, 0, stream>>>(lat, b2, ffo, NLAT);   // ffo = x + b2
    dim3 grid(NTOK / 16, 16);                                    // 16 chunk groups
    ffw_fused<<<grid, 32, 0, stream>>>(xnb, W1t, b1, W2t, ffo, (CFFH / 32) / 16);
    copyk<<<ew(NLAT), 256, 0, stream>>>(ffo, lat, NLAT);
  };

  // --- initial cross attention (no residual) ---
  cross_attn(lat);
  // --- initial transformer stack (no residual) ---
  self_attn(lat, 0);
  self_attn(lat, 1);

  // --- shared blocks ---
  for (int na = 0; na < 2; ++na) {
    cross_attn(prj);
    avg2<<<ew(NLAT), 256, 0, stream>>>(prj, lat, lat, NLAT);
    ffw(caff_g, caff_b, caW1_t, caff_b1, caW2_t, caff_b2);
    copyk<<<ew(NLAT), 256, 0, stream>>>(lat, zb, NLAT);
    self_attn(zb, 0);
    self_attn(zb, 1);
    avg2<<<ew(NLAT), 256, 0, stream>>>(zb, lat, lat, NLAT);
    ffw(trff_g, trff_b, trW1_t, trff_b1, trW2_t, trff_b2);
  }

  copyk<<<ew(NLAT), 256, 0, stream>>>(lat, (float*)d_out, NLAT);
}
